// MP_layer_dm_89481348645415
// MI455X (gfx1250) — matrix operand order repaired and checked by reading, not compiled, not run on hardware
//
#include <hip/hip_runtime.h>
#include <stdint.h>

// ---------------------------------------------------------------------------
// MP layer for MI455X (gfx1250, wave32, WMMA).
// Edge GEMM [E,144]@[144,64] in bf16 WMMA (fp32 acc), ELU, fixed-K=16 mean
// (one WMMA M-tile == one node's 16 edges), then node GEMM [N,128]@[128,64].
// ---------------------------------------------------------------------------

typedef __attribute__((ext_vector_type(16))) __bf16 v16bf;
typedef __attribute__((ext_vector_type(8)))  float  v8f;

#define D_NODE 64
#define D_EDGE 16
#define D_MESS 64
#define D_OUT  64
#define KDEG   16
#define NPB    8                 // nodes per block
#define EPB    (NPB * KDEG)      // 128 edges per block -> 8 M-tiles
#define KPSI   (2 * D_NODE + D_EDGE)   // 144
#define A_STRIDE 168             // LDS edge-row stride in bf16 (336B = 84 dw, conflict-free)
#define G_STRIDE 136             // LDS gamma-row stride in bf16 (272B)

__device__ __forceinline__ unsigned short f2bf(float f) {
  union { float f; unsigned u; } v; v.f = f;
  unsigned r = v.u + 0x7FFFu + ((v.u >> 16) & 1u);   // round-to-nearest-even
  return (unsigned short)(r >> 16);
}
__device__ __forceinline__ unsigned packbf(float a, float b) {
  return (unsigned)f2bf(a) | ((unsigned)f2bf(b) << 16);
}

// ---- prep: fp32 -> bf16 bulk convert (for x) -------------------------------
__global__ void k_cvt_bf16(const float* __restrict__ in,
                           unsigned* __restrict__ out, int npairs) {
  int i = blockIdx.x * blockDim.x + threadIdx.x;
  if (i < npairs) {
    float2 f = ((const float2*)in)[i];
    out[i] = packbf(f.x, f.y);
  }
}

// ---- prep: swizzle W_psi [144,64] into per-lane WMMA B fragments -----------
// Fragment (kt,nt): 32x16 B-tile. Lane layout (this code's contiguous k order): lane = {half,col};
// lane holds K = kt*32 + 16*half + (0..15), packed 2 bf16 per VGPR.
// Stored contiguously: frag[(kt*4+nt)][lane][16].  K padded 144 -> 160.
__global__ void k_pack_psi(const float* __restrict__ W,
                           unsigned short* __restrict__ out) {
  int t = blockIdx.x * blockDim.x + threadIdx.x;
  if (t >= 20 * 32) return;                 // 5 ktiles * 4 ntiles * 32 lanes
  int frag = t >> 5, lane = t & 31;
  int kt = frag >> 2, nt = frag & 3;
  int half = lane >> 4, n = nt * 16 + (lane & 15);
  unsigned short* o = out + t * 16;
  for (int i = 0; i < 16; ++i) {
    int k = kt * 32 + half * 16 + i;
    float v = (k < KPSI) ? W[k * D_MESS + n] : 0.f;
    o[i] = f2bf(v);
  }
}

// ---- prep: swizzle W_gamma [128,64] the same way ---------------------------
__global__ void k_pack_gamma(const float* __restrict__ W,
                             unsigned short* __restrict__ out) {
  int t = blockIdx.x * blockDim.x + threadIdx.x;
  if (t >= 16 * 32) return;                 // 4 ktiles * 4 ntiles * 32 lanes
  int frag = t >> 5, lane = t & 31;
  int kt = frag >> 2, nt = frag & 3;
  int half = lane >> 4, n = nt * 16 + (lane & 15);
  unsigned short* o = out + t * 16;
  for (int i = 0; i < 16; ++i) {
    int k = kt * 32 + half * 16 + i;        // always < 128
    o[i] = f2bf(W[k * D_OUT + n]);
  }
}

// ---- main fused kernel -----------------------------------------------------
__global__ __launch_bounds__(256) void k_mp(
    const unsigned short* __restrict__ xbf,   // [N,64] bf16
    const long long* __restrict__ ei,         // [2,E] int64
    const float* __restrict__ efeat,          // [E,16] f32
    const unsigned short* __restrict__ wpsi,  // swizzled B frags (20*32*16)
    const unsigned short* __restrict__ wg,    // swizzled B frags (16*32*16)
    const float* __restrict__ bpsi, const float* __restrict__ bg,
    float* __restrict__ out, long long E)
{
  __shared__ __align__(16) unsigned short sA[EPB * A_STRIDE];  // 43 KB edge A
  __shared__ __align__(16) unsigned short sG[16 * G_STRIDE];   // 4.25 KB gamma A
  __shared__ float sPart[NPB * 64 * 2];                        // 4 KB partials

  const int tid  = threadIdx.x;
  const int wave = tid >> 5, lane = tid & 31;
  const int half = lane >> 4, l15 = lane & 15;
  const int blk  = blockIdx.x;

  // ---- phase 0: gather+concat edge rows into LDS, bf16 ---------------------
  // 4 threads per edge, 64 edges per pass, 2 passes.
  #pragma unroll
  for (int p = 0; p < 2; ++p) {
    int le = p * 64 + (tid >> 2);
    int c  = tid & 3;
    long long ge = (long long)blk * EPB + le;
    int src = (int)ei[ge];
    int drn = (int)ei[E + ge];
    unsigned short* row = sA + le * A_STRIDE;
    const uint4* xs = (const uint4*)(xbf + (size_t)src * D_NODE + c * 16);
    ((uint4*)(row + c * 16))[0] = xs[0];
    ((uint4*)(row + c * 16))[1] = xs[1];
    const uint4* xd = (const uint4*)(xbf + (size_t)drn * D_NODE + c * 16);
    ((uint4*)(row + 64 + c * 16))[0] = xd[0];
    ((uint4*)(row + 64 + c * 16))[1] = xd[1];
    float4 ef = ((const float4*)(efeat + ge * D_EDGE))[c];
    uint2 ev; ev.x = packbf(ef.x, ef.y); ev.y = packbf(ef.z, ef.w);
    *(uint2*)(row + 128 + c * 4) = ev;
    uint2 z; z.x = 0u; z.y = 0u;                 // K pad 144..159 = 0
    *(uint2*)(row + 144 + c * 4) = z;
  }
  // stage this block's x rows (bf16) into gamma A-buffer cols 0..63
  if (tid < 64) {
    int node = tid >> 3, ch = tid & 7;
    ((uint4*)(sG + node * G_STRIDE + ch * 8))[0] =
        ((const uint4*)(xbf + ((size_t)blk * NPB + node) * D_NODE + ch * 8))[0];
  }
  __syncthreads();

  // ---- phase 1: edge GEMM. wave w owns node w (M-tile w), all 4 N-tiles ----
  v8f acc[4];
  #pragma unroll
  for (int i = 0; i < 4; ++i) acc[i] = (v8f){0.f,0.f,0.f,0.f,0.f,0.f,0.f,0.f};

  const unsigned short* aBase = sA + (wave * 16 + l15) * A_STRIDE + 16 * half;
  #pragma unroll
  for (int kt = 0; kt < 5; ++kt) {
    union { uint4 q[2]; v16bf v; } a;
    const unsigned short* ap = aBase + kt * 32;
    a.q[0] = *(const uint4*)ap;            // K = kt*32 + 16*half + 0..7
    a.q[1] = *(const uint4*)(ap + 8);      // K = kt*32 + 16*half + 8..15
    #pragma unroll
    for (int nt = 0; nt < 4; ++nt) {
      union { uint4 q[2]; v16bf v; } b;
      const unsigned short* bp = wpsi + (((kt * 4 + nt) * 32 + lane) << 4);
      b.q[0] = *(const uint4*)bp;
      b.q[1] = *(const uint4*)(bp + 8);
      acc[nt] = __builtin_amdgcn_wmma_f32_16x16x32_bf16(
          false, a.v, false, b.v, (short)0, acc[nt], false, false);
    }
  }

  // ---- phase 2: +bias, ELU, mean over the 16 edge-rows of this node --------
  // C layout: lane(half,col) VGPR v holds element (M = v + 8*half, N = col).
  #pragma unroll
  for (int nt = 0; nt < 4; ++nt) {
    float bias = bpsi[nt * 16 + l15];
    float s = 0.f;
    #pragma unroll
    for (int v = 0; v < 8; ++v) {
      float t = acc[nt][v] + bias;
      s += (t > 0.f) ? t : (__expf(t) - 1.f);
    }
    sPart[(wave * 64 + nt * 16 + l15) * 2 + half] = s;
  }
  __syncthreads();
  {
    int node = tid >> 5, cp = tid & 31;    // 8 nodes x 32 column-pairs
    int c0 = cp * 2;
    float m0 = (sPart[(node * 64 + c0    ) * 2] + sPart[(node * 64 + c0    ) * 2 + 1]) * (1.f / KDEG);
    float m1 = (sPart[(node * 64 + c0 + 1) * 2] + sPart[(node * 64 + c0 + 1) * 2 + 1]) * (1.f / KDEG);
    *(unsigned*)(sG + node * G_STRIDE + 64 + c0) = packbf(m0, m1);
  }
  __syncthreads();

  // ---- phase 3: gamma GEMM [16(8 valid) x 128] @ [128 x 64], waves 0..3 ----
  if (wave < 4) {
    int nt = wave;
    v8f g = (v8f){0.f,0.f,0.f,0.f,0.f,0.f,0.f,0.f};
    const unsigned short* gaBase = sG + l15 * G_STRIDE + 16 * half;
    #pragma unroll
    for (int kt = 0; kt < 4; ++kt) {
      union { uint4 q[2]; v16bf v; } a;
      const unsigned short* ap = gaBase + kt * 32;
      a.q[0] = *(const uint4*)ap;
      a.q[1] = *(const uint4*)(ap + 8);
      union { uint4 q[2]; v16bf v; } b;
      const unsigned short* bp = wg + (((kt * 4 + nt) * 32 + lane) << 4);
      b.q[0] = *(const uint4*)bp;
      b.q[1] = *(const uint4*)(bp + 8);
      g = __builtin_amdgcn_wmma_f32_16x16x32_bf16(
          false, a.v, false, b.v, (short)0, g, false, false);
    }
    if (half == 0) {                        // rows 0..7 = the 8 valid nodes
      float bias = bg[nt * 16 + l15];
      #pragma unroll
      for (int v = 0; v < 8; ++v)
        out[((size_t)blk * NPB + v) * D_OUT + nt * 16 + l15] = g[v] + bias;
    }
  }
}

// ---------------------------------------------------------------------------
extern "C" void kernel_launch(void* const* d_in, const int* in_sizes, int n_in,
                              void* d_out, int out_size, void* d_ws, size_t ws_size,
                              hipStream_t stream) {
  const float*     x  = (const float*)d_in[0];
  const long long* ei = (const long long*)d_in[1];
  const float*     e  = (const float*)d_in[2];
  const float*     Wp = (const float*)d_in[3];
  const float*     bp = (const float*)d_in[4];
  const float*     Wg = (const float*)d_in[5];
  const float*     bg = (const float*)d_in[6];
  float* out = (float*)d_out;

  int N = in_sizes[0] / D_NODE;                 // 50000
  long long E = (long long)in_sizes[2] / D_EDGE;

  char* ws = (char*)d_ws;
  unsigned short* xbf = (unsigned short*)ws;    // N*64 bf16 (~6.4 MB)
  size_t xbytes = (((size_t)N * D_NODE * 2) + 255) & ~(size_t)255;
  unsigned short* wpsiF = (unsigned short*)(ws + xbytes);           // 20 KB
  unsigned short* wgF   = (unsigned short*)(ws + xbytes + 20480);   // 16 KB

  int xpairs = N * D_NODE / 2;
  k_cvt_bf16<<<(xpairs + 255) / 256, 256, 0, stream>>>(x, (unsigned*)xbf, xpairs);
  k_pack_psi  <<<3, 256, 0, stream>>>(Wp, wpsiF);
  k_pack_gamma<<<2, 256, 0, stream>>>(Wg, wgF);
  k_mp<<<N / NPB, 256, 0, stream>>>(xbf, ei, e, wpsiF, wgF, bp, bg, out, E);
}
